// AttnGCNNet_56848187130261
// MI455X (gfx1250) — matrix operand order repaired and checked by reading, not compiled, not run on hardware
//
#include <hip/hip_runtime.h>
#include <math.h>

typedef __attribute__((ext_vector_type(16))) _Float16 v16h;
typedef __attribute__((ext_vector_type(8)))  _Float16 v8h;
typedef __attribute__((ext_vector_type(8)))  float    v8f;

#define NB     128
#define NPER   256
#define NNODE  (NB*NPER)        // 32768
#define NEDGE  262144
#define SEQL   1000
#define DMODEL 128

static __device__ __forceinline__ v8f wmma16(v16h a, v16h b, v8f c) {
  return __builtin_amdgcn_wmma_f32_16x16x32_f16(false, a, false, b, (short)0, c, false, false);
}

static __device__ __forceinline__ v16h mk16(v8h lo, v8h hi) {
  v16h r;
#pragma unroll
  for (int i = 0; i < 8; ++i) { r[i] = lo[i]; r[8 + i] = hi[i]; }
  return r;
}

// A fragment: 16 contiguous halves (two 16B loads at +0 and +8)
static __device__ __forceinline__ v16h loadA(const _Float16* p) {
  return mk16(*(const v8h*)(p), *(const v8h*)(p + 8));
}
// B fragment from packed layout: 16 contiguous halves (two 16B loads)
static __device__ __forceinline__ v16h loadB(const _Float16* p) {
  return mk16(*(const v8h*)(p), *(const v8h*)(p + 8));
}

// wave32 WMMA fragment lane roles used here (contiguous k per lane half, not the ISA table order):
//  A (16x32 f16): lane&15 = row; halves = K cols [kh..kh+15], kh=(lane>>4)*16
//  B (32x16 f16): lane&15 = col; halves = K rows [kb..kb+15], kb=(lane>>4)*16
//  C (16x16 f32): lane&15 = col; rows (lane>>4)*8 + r
#define WMMA_ROLES() \
  const int lane  = threadIdx.x & 31; \
  const int fa_m  = lane & 15; \
  const int fa_kh = (lane >> 4) * 16; \
  const int fb_n  = lane & 15; \
  const int fb_k  = (lane >> 4) * 16; \
  const int fc_n  = lane & 15; \
  const int fc_m  = (lane >> 4) * 8

// ---------------------------------------------------------------------------
// Packed-B GEMM, 4-wide N-blocking: one wave computes a 16x64 strip.
// A f16 row-major [M x lda] (K zero-padded), Bp fragment-packed
// f16 [K/32][Npad][32], Npad%64==0, K%32==0, M%16==0.
// Per chunk: 2 A loads + 8 B loads + 4 WMMAs.
// ---------------------------------------------------------------------------
__global__ __launch_bounds__(32)
void gemm16(const _Float16* __restrict__ A, int lda,
            const _Float16* __restrict__ Bp, int Npad,
            const float* __restrict__ bias,
            float* __restrict__ Cf, int ldcf,
            _Float16* __restrict__ Ch, int ldch,
            int K, int N, int relu)
{
  const int ngrp = Npad >> 6;
  const int row0 = (blockIdx.x / ngrp) << 4;
  const int col0 = (blockIdx.x % ngrp) << 6;
  WMMA_ROLES();
  const _Float16* arow = A + (size_t)(row0 + fa_m) * lda + fa_kh;
  const _Float16* bcol = Bp + ((size_t)(col0 + fb_n)) * 32 + fb_k;
  const size_t bstep = (size_t)Npad * 32;
  v8f acc[4] = {};
  for (int k0 = 0; k0 < K; k0 += 32) {
    v16h a = loadA(arow + k0);
#pragma unroll
    for (int j = 0; j < 4; ++j)
      acc[j] = wmma16(a, loadB(bcol + j * 512), acc[j]);
    bcol += bstep;
  }
#pragma unroll
  for (int j = 0; j < 4; ++j) {
    int n = col0 + j * 16 + fc_n;
    if (n < N) {
      float bv = bias ? bias[n] : 0.f;
#pragma unroll
      for (int r = 0; r < 8; ++r) {
        float v = acc[j][r] + bv;
        if (relu) v = fmaxf(v, 0.f);
        int m = row0 + fc_m + r;
        if (Cf) Cf[(size_t)m * ldcf + n] = v;
        if (Ch) Ch[(size_t)m * ldch + n] = (_Float16)v;
      }
    }
  }
}

// ---------------------------------------------------------------------------
// q = emb16[target] @ attn_in (packed, Npad=128). M=128000, K=128. f16 out.
// 4-wide N-blocking: grid = (M/16) * 2.
// ---------------------------------------------------------------------------
__global__ __launch_bounds__(32)
void embed_q16(const _Float16* __restrict__ emb16, const int* __restrict__ target,
               const _Float16* __restrict__ Bp, _Float16* __restrict__ q)
{
  const int row0 = (blockIdx.x >> 1) << 4;
  const int col0 = (blockIdx.x & 1) << 6;
  WMMA_ROLES();
  const _Float16* arow = emb16 + (size_t)target[row0 + fa_m] * DMODEL + fa_kh;
  const _Float16* bcol = Bp + ((size_t)(col0 + fb_n)) * 32 + fb_k;
  v8f acc[4] = {};
#pragma unroll
  for (int k0 = 0; k0 < DMODEL; k0 += 32) {
    v16h a = loadA(arow + k0);
#pragma unroll
    for (int j = 0; j < 4; ++j)
      acc[j] = wmma16(a, loadB(bcol + j * 512), acc[j]);
    bcol += (size_t)DMODEL * 32;
  }
#pragma unroll
  for (int j = 0; j < 4; ++j)
#pragma unroll
    for (int r = 0; r < 8; ++r)
      q[(size_t)(row0 + fc_m + r) * DMODEL + col0 + j * 16 + fc_n] = (_Float16)acc[j][r];
}

// ---------------------------------------------------------------------------
// Fused attention per (b, 16-query tile): scores -> softmax -> mix -> out-proj.
// All matmul stages use 4-wide tile groups (4 accumulators per group).
// ---------------------------------------------------------------------------
__global__ __launch_bounds__(32)
void attn_fused(const _Float16* __restrict__ q,
                const _Float16* __restrict__ ctxTp,
                const _Float16* __restrict__ ctxp,
                const _Float16* __restrict__ Woutp,
                _Float16* __restrict__ att)
{
  __shared__ _Float16 qs[16][DMODEL];
  __shared__ float    sc[16][NPER];
  __shared__ _Float16 ps[16][NPER];
  __shared__ _Float16 mixs[16][DMODEL];

  const int b   = blockIdx.x;
  const int qt  = blockIdx.y;
  const int tid = threadIdx.x;
  WMMA_ROLES();
  const _Float16* ctxT_b = ctxTp + (size_t)b * (NPER * DMODEL);
  const _Float16* ctx_b  = ctxp  + (size_t)b * (NPER * DMODEL);

  // stage q tile to LDS
  for (int i = tid; i < 16 * DMODEL / 8; i += 32) {
    int m = i >> 4, d = (i & 15) * 8;
    int s = qt * 16 + m;
    v8h v = {};
    if (s < SEQL) v = *(const v8h*)(q + ((size_t)b * SEQL + s) * DMODEL + d);
    *(v8h*)&qs[m][d] = v;
  }
  __syncthreads();

  // scores: S[16x256] = q_tile @ ctx^T  (K=128, Npad=256; groups of 4 k-tiles)
  {
    v16h aq[4];
#pragma unroll
    for (int dc = 0; dc < 4; ++dc)
      aq[dc] = loadA(&qs[fa_m][dc * 32 + fa_kh]);
    for (int ktg = 0; ktg < 4; ++ktg) {
      const _Float16* bbase = ctxT_b + ((size_t)(ktg * 64 + fb_n)) * 32 + fb_k;
      v8f acc[4] = {};
#pragma unroll
      for (int dc = 0; dc < 4; ++dc) {
        const _Float16* bc = bbase + (size_t)dc * (NPER * 32);
#pragma unroll
        for (int j = 0; j < 4; ++j)
          acc[j] = wmma16(aq[dc], loadB(bc + j * 512), acc[j]);
      }
#pragma unroll
      for (int j = 0; j < 4; ++j)
#pragma unroll
        for (int r = 0; r < 8; ++r)
          sc[fc_m + r][ktg * 64 + j * 16 + fc_n] = acc[j][r];
    }
  }
  __syncthreads();

  // row softmax (thread per row), probs to f16 LDS
  if (tid < 16) {
    float mx = -1e30f;
    for (int j = 0; j < NPER; ++j) mx = fmaxf(mx, sc[tid][j]);
    float sum = 0.f;
    for (int j = 0; j < NPER; ++j) { float e = __expf(sc[tid][j] - mx); sc[tid][j] = e; sum += e; }
    float inv = 1.f / sum;
    for (int j = 0; j < NPER; ++j) ps[tid][j] = (_Float16)(sc[tid][j] * inv);
  }
  __syncthreads();

  // mix = P[16x256] @ ctx[256x128]  (K=256, Npad=128; 2 groups of 4 n-tiles)
  for (int ntg = 0; ntg < 2; ++ntg) {
    const _Float16* bbase = ctx_b + ((size_t)(ntg * 64 + fb_n)) * 32 + fb_k;
    v8f acc[4] = {};
#pragma unroll
    for (int kc = 0; kc < 8; ++kc) {
      v16h a = loadA(&ps[fa_m][kc * 32 + fa_kh]);
      const _Float16* bc = bbase + (size_t)kc * (DMODEL * 32);
#pragma unroll
      for (int j = 0; j < 4; ++j)
        acc[j] = wmma16(a, loadB(bc + j * 512), acc[j]);
    }
#pragma unroll
    for (int j = 0; j < 4; ++j)
#pragma unroll
      for (int r = 0; r < 8; ++r)
        mixs[fc_m + r][ntg * 64 + j * 16 + fc_n] = (_Float16)acc[j][r];
  }
  __syncthreads();

  // att = tanh(cat(mix, q) @ Wout[256x128])  (chunks 0..3 mixs, 4..7 qs)
  for (int ntg = 0; ntg < 2; ++ntg) {
    const _Float16* bbase = Woutp + ((size_t)(ntg * 64 + fb_n)) * 32 + fb_k;
    v8f acc[4] = {};
#pragma unroll
    for (int kc = 0; kc < 8; ++kc) {
      const _Float16* asrc = (kc < 4) ? &mixs[fa_m][kc * 32 + fa_kh]
                                      : &qs[fa_m][(kc - 4) * 32 + fa_kh];
      v16h a = loadA(asrc);
      const _Float16* bc = bbase + (size_t)kc * (DMODEL * 32);
#pragma unroll
      for (int j = 0; j < 4; ++j)
        acc[j] = wmma16(a, loadB(bc + j * 512), acc[j]);
    }
#pragma unroll
    for (int j = 0; j < 4; ++j)
#pragma unroll
      for (int r = 0; r < 8; ++r) {
        int s = qt * 16 + fc_m + r;
        if (s < SEQL)
          att[((size_t)b * SEQL + s) * DMODEL + ntg * 64 + j * 16 + fc_n] =
              (_Float16)tanhf(acc[j][r]);
      }
  }
}

// ---------------------------------------------------------------------------
// Conv1d as GEMM per batch: out[b][o][p] = sum_{c,t} w16[o][c*8+t] * att[b][c][p+t]
// A = conv_w f16 [32 x 8000] row-major, K=8000, N=121. f16 out (o*121+p).
// ---------------------------------------------------------------------------
__global__ __launch_bounds__(32)
void conv_wmma16(const _Float16* __restrict__ att, const _Float16* __restrict__ W16,
                 const float* __restrict__ cb, _Float16* __restrict__ out)
{
  const int b    = blockIdx.x;
  const int row0 = (int)blockIdx.y << 4;   // output channel tile (0,16)
  const int col0 = (int)blockIdx.z << 4;   // position tile (0..7)
  WMMA_ROLES();
  const _Float16* attb = att + (size_t)b * SEQL * DMODEL;
  const _Float16* arow = W16 + (size_t)(row0 + fa_m) * 8000 + fa_kh;
  const int p = col0 + fb_n;
  const int pok = (p < 121);
  v8f acc = {};
  for (int k0 = 0; k0 < 8000; k0 += 32) {
    v16h a = loadA(arow + k0);
    v16h bfr;
    if (pok) {
      const int c0 = (k0 + fb_k) >> 3;           // two consecutive channels
      const _Float16* r0 = attb + (size_t)c0 * DMODEL + p;
      const _Float16* r1 = r0 + DMODEL;
#pragma unroll
      for (int i = 0; i < 8; ++i) { bfr[i] = r0[i]; bfr[8 + i] = r1[i]; }
    } else {
#pragma unroll
      for (int i = 0; i < 16; ++i) bfr[i] = (_Float16)0.f;
    }
    acc = wmma16(a, bfr, acc);
  }
#pragma unroll
  for (int r = 0; r < 8; ++r) {
    int o = row0 + fc_m + r, pp = col0 + fc_n;
    if (pp < 121)
      out[(size_t)b * (32 * 121) + o * 121 + pp] = (_Float16)(acc[r] + cb[o]);
  }
}

// --------------------------- pack / convert helpers ------------------------
// Fragment-pack f32 weight [K x N] row-major -> f16 [Kpad/32][Npad][32], zero pad.
__global__ void pack_bw(const float* __restrict__ src, int K, int N, int Npad,
                        _Float16* __restrict__ out, int total)
{
  int idx = blockIdx.x * 256 + threadIdx.x;
  if (idx >= total) return;
  int c = idx / (Npad * 32);
  int rem = idx - c * (Npad * 32);
  int n = rem >> 5, w = rem & 31;
  int k = c * 32 + w;
  out[idx] = (k < K && n < N) ? (_Float16)src[(size_t)k * N + n] : (_Float16)0.f;
}

// Batched fragment-pack from f16 src [b][K x lds] (optional transpose).
__global__ void pack_b16(const _Float16* __restrict__ src, int lds,
                         int K, int Npad, int trans,
                         _Float16* __restrict__ out,
                         int sstride, int ostride, int total)
{
  int b = blockIdx.y;
  int idx = blockIdx.x * 256 + threadIdx.x;
  if (idx >= total) return;
  int c = idx / (Npad * 32);
  int rem = idx - c * (Npad * 32);
  int n = rem >> 5, w = rem & 31;
  int k = c * 32 + w;
  const _Float16* s = src + (size_t)b * sstride;
  _Float16 v = (k < K) ? (trans ? s[(size_t)n * lds + k] : s[(size_t)k * lds + n])
                       : (_Float16)0.f;
  out[(size_t)b * ostride + idx] = v;
}

__global__ void cvt16(const float* __restrict__ src, _Float16* __restrict__ dst, int n) {
  int i = blockIdx.x * 256 + threadIdx.x;
  if (i < n) dst[i] = (_Float16)src[i];
}

__global__ void pad_x16(const float* __restrict__ x, _Float16* __restrict__ xp) {
  int n = blockIdx.x, d = threadIdx.x;   // 128 threads
  xp[(size_t)n * DMODEL + d] = (d < 78) ? (_Float16)x[(size_t)n * 78 + d] : (_Float16)0.f;
}

// --------------------------- graph helpers ---------------------------------
__global__ void deg_kernel(const int* __restrict__ dst, float* __restrict__ deg) {
  int e = blockIdx.x * blockDim.x + threadIdx.x;
  if (e < NEDGE) atomicAdd(&deg[dst[e]], 1.f);
}

__global__ void dis_kernel(const float* __restrict__ deg, float* __restrict__ dis) {
  int i = blockIdx.x * blockDim.x + threadIdx.x;
  if (i < NNODE) dis[i] = rsqrtf(1.f + deg[i]);
}

__global__ void edge_scatter(const int* __restrict__ src, const int* __restrict__ dst,
                             const float* __restrict__ dis, const float* __restrict__ hW,
                             float* __restrict__ agg, int dim)
{
  int e = blockIdx.x;
  int d = threadIdx.x;
  if (d >= dim) return;
  int s = src[e], t = dst[e];
  float w = dis[s] * dis[t];
  atomicAdd(&agg[(size_t)t * DMODEL + d], hW[(size_t)s * DMODEL + d] * w);
}

// out = relu(agg + hW*dis^2 + b) (+ zero-padded residual), f16 with zero
// padding of columns [dim, 128) so downstream GEMMs need no guards.
__global__ void gcn_finish(const float* __restrict__ agg, const float* __restrict__ hWf,
                           const float* __restrict__ dis, const float* __restrict__ bias,
                           const _Float16* __restrict__ res, int res_dim,
                           _Float16* __restrict__ out, int dim)
{
  int n = blockIdx.x;
  int d = threadIdx.x;    // 128 threads
  _Float16 o = (_Float16)0.f;
  if (d < dim) {
    float sn = dis[n]; sn *= sn;
    float v = agg[(size_t)n * DMODEL + d] + hWf[(size_t)n * DMODEL + d] * sn + bias[d];
    v = fmaxf(v, 0.f);
    if (res && d < res_dim) v += (float)res[(size_t)n * DMODEL + d];
    o = (_Float16)v;
  }
  out[(size_t)n * DMODEL + d] = o;
}

__global__ void pool_max16(const _Float16* __restrict__ h, _Float16* __restrict__ g) {
  int b = blockIdx.x, d = threadIdx.x;
  float m = -1e30f;
  for (int n = 0; n < NPER; ++n)
    m = fmaxf(m, (float)h[((size_t)b * NPER + n) * DMODEL + d]);
  g[(size_t)b * DMODEL + d] = (_Float16)m;
}

__global__ void out_kernel(const float* __restrict__ xc2, const float* __restrict__ w,
                           const float* __restrict__ b0, float* __restrict__ out) {
  int b = threadIdx.x;
  float s = 0.f;
  for (int j = 0; j < 512; ++j) s += xc2[(size_t)b * 512 + j] * w[j];
  out[b] = s + b0[0];
}

// ---------------------------------------------------------------------------
extern "C" void kernel_launch(void* const* d_in, const int* in_sizes, int n_in,
                              void* d_out, int out_size, void* d_ws, size_t ws_size,
                              hipStream_t stream) {
  (void)in_sizes; (void)n_in; (void)out_size; (void)ws_size;
  const float* x      = (const float*)d_in[0];
  const int*   eidx   = (const int*)d_in[1];
  const int*   esrc   = eidx;
  const int*   edst   = eidx + NEDGE;
  const int*   target = (const int*)d_in[3];
  const float* Ws[6]  = {(const float*)d_in[4],  (const float*)d_in[6],
                         (const float*)d_in[8],  (const float*)d_in[10],
                         (const float*)d_in[12], (const float*)d_in[14]};
  const float* bsv[6] = {(const float*)d_in[5],  (const float*)d_in[7],
                         (const float*)d_in[9],  (const float*)d_in[11],
                         (const float*)d_in[13], (const float*)d_in[15]};
  const float* fcg1_w = (const float*)d_in[16]; const float* fcg1_b = (const float*)d_in[17];
  const float* fcg2_w = (const float*)d_in[18]; const float* fcg2_b = (const float*)d_in[19];
  const float* emb    = (const float*)d_in[20];
  const float* attn_in_w  = (const float*)d_in[21];
  const float* attn_out_w = (const float*)d_in[22];
  const float* conv_w = (const float*)d_in[23]; const float* conv_b = (const float*)d_in[24];
  const float* fc1xt_w = (const float*)d_in[25]; const float* fc1xt_b = (const float*)d_in[26];
  const float* fc1_w = (const float*)d_in[27]; const float* fc1_b = (const float*)d_in[28];
  const float* fc2_w = (const float*)d_in[29]; const float* fc2_b = (const float*)d_in[30];
  const float* out_w = (const float*)d_in[31]; const float* out_b = (const float*)d_in[32];

  // ---- workspace (bump allocator, 256B aligned) ----
  char* wp = (char*)d_ws;
  auto alloc = [&](size_t bytes) { void* r = wp; wp += (bytes + 255) & ~(size_t)255; return r; };

  float* deg  = (float*)alloc(NNODE * 4);
  float* dis  = (float*)alloc(NNODE * 4);
  float* hW   = (float*)alloc((size_t)NNODE * DMODEL * 4);
  float* agg  = (float*)alloc((size_t)NNODE * DMODEL * 4);
  _Float16* xp  = (_Float16*)alloc((size_t)NNODE * DMODEL * 2);
  _Float16* h0h = (_Float16*)alloc((size_t)NNODE * DMODEL * 2);
  _Float16* h1h = (_Float16*)alloc((size_t)NNODE * DMODEL * 2);
  _Float16* qh   = (_Float16*)alloc((size_t)NB * SEQL * DMODEL * 2);
  _Float16* atth = (_Float16*)alloc((size_t)NB * SEQL * DMODEL * 2);
  _Float16* ctxTp = (_Float16*)alloc((size_t)NB * NPER * DMODEL * 2);
  _Float16* ctxp  = (_Float16*)alloc((size_t)NB * NPER * DMODEL * 2);
  _Float16* conv_out16 = (_Float16*)alloc((size_t)NB * 32 * 121 * 2);
  _Float16* g16  = (_Float16*)alloc((size_t)NB * DMODEL * 2);
  _Float16* xg1h = (_Float16*)alloc((size_t)NB * 1024 * 2);
  _Float16* xch  = (_Float16*)alloc((size_t)NB * 256 * 2);
  _Float16* xc1h = (_Float16*)alloc((size_t)NB * 1024 * 2);
  float*    xc2  = (float*)alloc((size_t)NB * 512 * 4);
  _Float16* emb16   = (_Float16*)alloc((size_t)26 * DMODEL * 2);
  _Float16* convw16 = (_Float16*)alloc((size_t)32 * 8000 * 2);

  auto ceil32 = [](int v) { return (v + 31) & ~31; };
  auto ceil64 = [](int v) { return (v + 63) & ~63; };

  // Npad rounded to 64 so every GEMM runs the 4-wide path.
  auto pack_w = [&](const float* src, int K, int N) {
    int Kp = ceil32(K), Np = ceil64(N);
    int total = Kp * Np;
    _Float16* dst = (_Float16*)alloc((size_t)total * 2);
    pack_bw<<<dim3((total + 255) / 256), dim3(256), 0, stream>>>(src, K, N, Np, dst, total);
    return dst;
  };

  // ---- pack all B operands ----
  const int kin[6]  = {78, 78, 88, 88, 118, 118};
  const int kout[6] = {78, 88, 88, 118, 118, 128};
  const int rdim[6] = {0,  78, 0,  88,  0,   0};
  _Float16* Wp[6];
  for (int l = 0; l < 6; ++l) Wp[l] = pack_w(Ws[l], kin[l], kout[l]);
  _Float16* fcg1p  = pack_w(fcg1_w, 128, 1024);
  _Float16* fcg2p  = pack_w(fcg2_w, 1024, 128);
  _Float16* fc1xtp = pack_w(fc1xt_w, 3872, 128);
  _Float16* fc1p   = pack_w(fc1_w, 256, 1024);
  _Float16* fc2p   = pack_w(fc2_w, 1024, 512);
  _Float16* ainp   = pack_w(attn_in_w, 128, 128);
  _Float16* aoutp  = pack_w(attn_out_w, 256, 128);
  cvt16<<<dim3((26 * DMODEL + 255) / 256), dim3(256), 0, stream>>>(emb, emb16, 26 * DMODEL);
  cvt16<<<dim3((32 * 8000 + 255) / 256), dim3(256), 0, stream>>>(conv_w, convw16, 32 * 8000);
  pad_x16<<<dim3(NNODE), dim3(DMODEL), 0, stream>>>(x, xp);

  // ---- degree / symmetric normalization ----
  hipMemsetAsync(deg, 0, NNODE * 4, stream);
  deg_kernel<<<dim3(NEDGE / 256), dim3(256), 0, stream>>>(edst, deg);
  dis_kernel<<<dim3(NNODE / 256), dim3(256), 0, stream>>>(deg, dis);

  // ---- six GCN layers (ping-pong f16 h buffers, ld = 128) ----
  const _Float16* hin = xp;
  _Float16* bufs[2] = {h0h, h1h};
  int cur = 0;
  for (int l = 0; l < 6; ++l) {
    _Float16* hout = bufs[cur];
    int Kp = ceil32(kin[l]), Np = ceil64(kout[l]);
    gemm16<<<dim3((NNODE / 16) * (Np / 64)), dim3(32), 0, stream>>>(
        hin, DMODEL, Wp[l], Np, nullptr, hW, DMODEL, nullptr, 0, Kp, kout[l], 0);
    hipMemsetAsync(agg, 0, (size_t)NNODE * DMODEL * 4, stream);
    edge_scatter<<<dim3(NEDGE), dim3(DMODEL), 0, stream>>>(esrc, edst, dis, hW, agg, kout[l]);
    const _Float16* res = rdim[l] ? hin : nullptr;
    gcn_finish<<<dim3(NNODE), dim3(DMODEL), 0, stream>>>(agg, hW, dis, bsv[l], res, rdim[l],
                                                         hout, kout[l]);
    hin = hout;
    cur ^= 1;
  }
  const _Float16* ctx = hin;   // final h f16: [B, 256, 128]

  // ---- batched ctx packs for attention ----
  {
    int total = NPER * DMODEL;   // per-b packed elements (Kpad*Npad)
    // scores B = ctx^T: K=128 (d), Npad=256 (node), transposed read
    pack_b16<<<dim3((total + 255) / 256, NB), dim3(256), 0, stream>>>(
        ctx, DMODEL, DMODEL, NPER, 1, ctxTp, NPER * DMODEL, total, total);
    // mix B = ctx: K=256 (node), Npad=128 (d)
    pack_b16<<<dim3((total + 255) / 256, NB), dim3(256), 0, stream>>>(
        ctx, DMODEL, NPER, DMODEL, 0, ctxp, NPER * DMODEL, total, total);
  }

  // ---- q projection, fused attention, conv ----
  embed_q16<<<dim3((NB * SEQL / 16) * 2), dim3(32), 0, stream>>>(emb16, target, ainp, qh);
  attn_fused<<<dim3(NB, (SEQL + 15) / 16), dim3(32), 0, stream>>>(qh, ctxTp, ctxp, aoutp, atth);
  conv_wmma16<<<dim3(NB, 2, 8), dim3(32), 0, stream>>>(atth, convw16, conv_b, conv_out16);

  // ---- graph branch: max-pool + MLP -> xc[:, 0:128] ----
  pool_max16<<<dim3(NB), dim3(DMODEL), 0, stream>>>(ctx, g16);
  gemm16<<<dim3((NB / 16) * (1024 / 64)), dim3(32), 0, stream>>>(
      g16, DMODEL, fcg1p, 1024, fcg1_b, nullptr, 0, xg1h, 1024, 128, 1024, 1);
  gemm16<<<dim3((NB / 16) * (128 / 64)), dim3(32), 0, stream>>>(
      xg1h, 1024, fcg2p, 128, fcg2_b, nullptr, 0, xch, 256, 1024, 128, 0);

  // ---- sequence branch: fc1xt -> xc[:, 128:256] ----
  gemm16<<<dim3((NB / 16) * (128 / 64)), dim3(32), 0, stream>>>(
      conv_out16, 32 * 121, fc1xtp, 128, fc1xt_b, nullptr, 0, xch + DMODEL, 256, 3872, 128, 0);

  // ---- head MLP ----
  gemm16<<<dim3((NB / 16) * (1024 / 64)), dim3(32), 0, stream>>>(
      xch, 256, fc1p, 1024, fc1_b, nullptr, 0, xc1h, 1024, 256, 1024, 1);
  gemm16<<<dim3((NB / 16) * (512 / 64)), dim3(32), 0, stream>>>(
      xc1h, 1024, fc2p, 512, fc2_b, xc2, 512, nullptr, 0, 1024, 512, 1);
  out_kernel<<<dim3(1), dim3(NB), 0, stream>>>(xc2, out_w, out_b, (float*)d_out);
}
